// InfoNCELoss_31224412242721
// MI455X (gfx1250) — matrix operand order repaired and checked by reading, not compiled, not run on hardware
//
#include <hip/hip_runtime.h>
#include <hip/hip_bf16.h>

typedef __attribute__((ext_vector_type(16))) _Float16 v16h;
typedef __attribute__((ext_vector_type(8)))  _Float16 v8h;
typedef __attribute__((ext_vector_type(8)))  float    v8f;

#define N2      8192      // total rows (2N)
#define NHALF   4096
#define DIM     1024
#define INVT    2.0f      // 1 / temperature
#define BM      128       // rows per workgroup (8 waves x 16)
#define JCHUNKS 4
#define TILES_PER_CHUNK ((N2 / 16) / JCHUNKS)   // 128 column tiles per chunk
#define BSTRIDE 1032      // padded LDS row stride in halfs (516 dwords == 4 mod 64)
#define TILE_HALFS (16 * BSTRIDE)               // one B tile in LDS (halfs)
#define SMEM_BYTES (2 * TILE_HALFS * 2)         // double-buffered, 66048 B

// ---- fp32 -> fp16 one-time conversion of F into workspace (L2-resident) ----
__global__ void infonce_prep(const float* __restrict__ F, _Float16* __restrict__ H) {
    size_t i = ((size_t)blockIdx.x * 256 + threadIdx.x) * 8;
    float4 f0 = *(const float4*)(F + i);
    float4 f1 = *(const float4*)(F + i + 4);
    v8h h;
    h[0] = (_Float16)f0.x; h[1] = (_Float16)f0.y;
    h[2] = (_Float16)f0.z; h[3] = (_Float16)f0.w;
    h[4] = (_Float16)f1.x; h[5] = (_Float16)f1.y;
    h[6] = (_Float16)f1.z; h[7] = (_Float16)f1.w;
    *(v8h*)(H + i) = h;
}

__global__ void infonce_zero_ws(float* ws) {
    int i = blockIdx.x * blockDim.x + threadIdx.x;
    if (i < 2 * N2) ws[i] = 0.0f;
}

// Issue one 16x1024 fp16 tile (32 KB) as 8 async 16-B copies per thread.
// GVS mode: mem_addr = SADDR + VADDR(32b) ; VDST = LDS byte address.
__device__ __forceinline__ void async_copy_tile(const _Float16* src_base,
                                                unsigned lds_base_bytes, int tid) {
    unsigned long long sbase = (unsigned long long)(size_t)src_base;
#pragma unroll
    for (int i = 0; i < 8; ++i) {
        const int c     = tid + 256 * i;     // 16-B chunk id, 128 per row
        const int row   = c >> 7;
        const int col16 = c & 127;
        unsigned lds_off = lds_base_bytes + (unsigned)(row * (BSTRIDE * 2) + col16 * 16);
        unsigned goff    = (unsigned)((row * DIM + col16 * 8) * 2);
        asm volatile("global_load_async_to_lds_b128 %0, %1, %2"
                     :: "v"(lds_off), "v"(goff), "s"(sbase)
                     : "memory");
    }
}

#define LDB(bp, kk)                                                             \
    ({ v8h _lo = *(const v8h*)((bp) + (kk) * 32);                               \
       v8h _hi = *(const v8h*)((bp) + (kk) * 32 + 8);                           \
       __builtin_shufflevector(_lo, _hi, 0,1,2,3,4,5,6,7,8,9,10,11,12,13,14,15); })

__global__ __launch_bounds__(256, 1)
void infonce_main(const _Float16* __restrict__ H,
                  float* __restrict__ ws_l,   // per-row sum of exp(logit)
                  float* __restrict__ ws_t) { // per-row target logit
    extern __shared__ _Float16 Bs[];          // [2][16][BSTRIDE]

    const int tid  = threadIdx.x;
    const int lane = tid & 31;
    const int wave = tid >> 5;
    const int n    = lane & 15;   // WMMA B/C column within tile
    const int kh   = lane >> 4;   // half-wave selector
    const int rb   = blockIdx.x;  // row block
    const int jc   = blockIdx.y;  // column chunk
    const int rowbase = rb * BM + wave * 16;

    const unsigned smem_base = (unsigned)(size_t)(const void*)Bs;  // LDS byte addr

    const int jt0   = jc * TILES_PER_CHUNK;
    const int jtEnd = jt0 + TILES_PER_CHUNK;

    // Kick off async staging of the first B tile before doing anything else.
    async_copy_tile(H + (size_t)(jt0 * 16) * DIM, smem_base, tid);

    // ---- Load this wave's 16x1024 A strip into WMMA A-operand registers ----
    // A layout (16-bit, 16x32): lane L holds row M=L%16;
    //   elems 0..15 -> k = 32*kk + 16*kh + {0..15}
    const _Float16* Arow = H + (size_t)(rowbase + n) * DIM + 16 * kh;
    v16h a[32];
#pragma unroll
    for (int kk = 0; kk < 32; ++kk) {
        v8h lo = *(const v8h*)(Arow + kk * 32);
        v8h hi = *(const v8h*)(Arow + kk * 32 + 8);
        a[kk] = __builtin_shufflevector(lo, hi, 0,1,2,3,4,5,6,7,8,9,10,11,12,13,14,15);
    }

    float l[8], t[8];
#pragma unroll
    for (int r = 0; r < 8; ++r) { l[r] = 0.0f; t[r] = 0.0f; }

    for (int jt = jt0; jt < jtEnd; ++jt) {
        const int cur = (jt - jt0) & 1;
        const int jb  = jt * 16;

        // My async copies for tile `cur` are done; barrier makes everyone's visible.
        asm volatile("s_wait_asynccnt 0x0" ::: "memory");
        __syncthreads();

        // Prefetch next tile into the other buffer while we compute this one.
        if (jt + 1 < jtEnd)
            async_copy_tile(H + (size_t)((jt + 1) * 16) * DIM,
                            smem_base + (1 - cur) * (TILE_HALFS * 2), tid);

        // ---- 16x16 logits tile: 32 WMMAs along K=1024 ----
        // B layout (32x16): lane L holds column n=L%16, k = 16*kh + {0..15}
        const _Float16* bp = Bs + cur * TILE_HALFS + n * BSTRIDE + kh * 16;
        v8f acc0 = {}, acc1 = {};
#pragma unroll
        for (int kk = 0; kk < 32; ++kk) {
            v16h b = LDB(bp, kk);
            if (kk & 1)
                acc1 = __builtin_amdgcn_wmma_f32_16x16x32_f16(
                           false, a[kk], false, b, (short)0, acc1, false, false);
            else
                acc0 = __builtin_amdgcn_wmma_f32_16x16x32_f16(
                           false, a[kk], false, b, (short)0, acc0, false, false);
        }

        // Depth-3 software pipeline: [6 ds-reads] ([1 wmma][2 ds-reads]) x29 [3 wmma]
        // 0x100 = DS read, 0x008 = MFMA/WMMA.
        __builtin_amdgcn_sched_group_barrier(0x100, 6, 0);
#pragma unroll
        for (int s = 0; s < 29; ++s) {
            __builtin_amdgcn_sched_group_barrier(0x008, 1, 0);
            __builtin_amdgcn_sched_group_barrier(0x100, 2, 0);
        }
        __builtin_amdgcn_sched_group_barrier(0x008, 3, 0);

        // ---- fused online softmax update (logits bounded by 1/T: no max) ----
        const int col = jb + n;
#pragma unroll
        for (int r = 0; r < 8; ++r) {
            const int row = rowbase + 8 * kh + r;   // C layout: M = r + 8*kh
            const float s = (acc0[r] + acc1[r]) * INVT;
            t[r] += (col == ((row + NHALF) & (N2 - 1))) ? s : 0.0f;
            l[r] += (col == row) ? 0.0f : __expf(s);
        }

        // All waves done reading buf[cur] before anyone overwrites it next iter.
        __syncthreads();
    }

    // ---- segmented reduction across the 16 lanes of each half-wave ----
#pragma unroll
    for (int r = 0; r < 8; ++r) {
        float lv = l[r], tv = t[r];
#pragma unroll
        for (int mk = 1; mk < 16; mk <<= 1) {
            lv += __shfl_xor(lv, mk, 32);
            tv += __shfl_xor(tv, mk, 32);
        }
        if (n == 0) {
            const int row = rowbase + 8 * kh + r;
            atomicAdd(&ws_l[row], lv);
            atomicAdd(&ws_t[row], tv);
        }
    }
}

__global__ void infonce_final(const float* __restrict__ ws_l,
                              const float* __restrict__ ws_t,
                              float* __restrict__ out) {
    const int tid = threadIdx.x;
    float s = 0.0f;
    for (int i = tid; i < N2; i += 256)
        s += logf(ws_l[i]) - ws_t[i];     // -(t - logsumexp)
#pragma unroll
    for (int mk = 1; mk < 32; mk <<= 1) s += __shfl_xor(s, mk, 32);
    __shared__ float red[8];
    if ((tid & 31) == 0) red[tid >> 5] = s;
    __syncthreads();
    if (tid == 0) {
        float tot = 0.0f;
        for (int w = 0; w < 8; ++w) tot += red[w];
        out[0] = tot / (float)N2;
    }
}

extern "C" void kernel_launch(void* const* d_in, const int* in_sizes, int n_in,
                              void* d_out, int out_size, void* d_ws, size_t ws_size,
                              hipStream_t stream) {
    const float* F = (const float*)d_in[0];
    float* out = (float*)d_out;

    // Workspace layout: fp16 copy of F (16 MB), then 2*8192 floats of reductions.
    _Float16* H    = (_Float16*)d_ws;
    float*    ws_l = (float*)((char*)d_ws + (size_t)N2 * DIM * sizeof(_Float16));
    float*    ws_t = ws_l + N2;

    infonce_prep<<<dim3((N2 * DIM) / 8 / 256), 256, 0, stream>>>(F, H);
    infonce_zero_ws<<<dim3((2 * N2 + 255) / 256), 256, 0, stream>>>(ws_l);

    dim3 grid(N2 / BM, JCHUNKS);
    infonce_main<<<grid, 256, SMEM_BYTES, stream>>>(H, ws_l, ws_t);

    infonce_final<<<1, 256, 0, stream>>>(ws_l, ws_t, out);
}
